// TrojanGNN_7696581394564
// MI455X (gfx1250) — compile-verified, matrix operand order checked by reading, not run on hardware
//
#include <hip/hip_runtime.h>
#include <hip/hip_bf16.h>

typedef __attribute__((ext_vector_type(16))) _Float16 v16h;
typedef __attribute__((ext_vector_type(8)))  float    v8f;

#define NNODES 50000
#define NEDGES 800000
#define DH     128
#define NCLS   16
#define BN_EPS 1e-5f

// ---------------- utility ----------------
__global__ void zero_k(float* __restrict__ p, size_t n) {
  size_t i = (size_t)blockIdx.x * blockDim.x + threadIdx.x;
  size_t stride = (size_t)gridDim.x * blockDim.x;
  for (; i < n; i += stride) p[i] = 0.0f;
}

__global__ void degree_k(const int* __restrict__ dst, float* __restrict__ deg, int E) {
  int i = blockIdx.x * blockDim.x + threadIdx.x;
  int stride = gridDim.x * blockDim.x;
  for (; i < E; i += stride) atomicAdd(&deg[dst[i]], 1.0f);
}

__global__ void rsqrt_k(float* __restrict__ d, int n) {
  int i = blockIdx.x * blockDim.x + threadIdx.x;
  int stride = gridDim.x * blockDim.x;
  for (; i < n; i += stride) d[i] = rsqrtf(1.0f + d[i]);
}

// ---------------- WMMA GEMM: C[N,OUT] = A[N,128] x W[128,OUT] ----------------
// block = 256 threads = 8 waves; each wave owns one 16-row tile.
// W is staged into LDS PRE-SWIZZLED into WMMA B-fragment order so each lane
// reads its 16 halves as one contiguous 32-byte LDS access (2 x ds_load_b128).
template <int OUT>
__global__ void gemm_wmma_k(const float* __restrict__ A, const float* __restrict__ W,
                            float* __restrict__ C, int N) {
  constexpr int KSTEPS = DH / 32;                  // 4
  constexpr int NTILES = OUT / 16;
  __shared__ __align__(32) _Float16 Wsw[NTILES * KSTEPS * 32 * 16];

  const int tid = threadIdx.x;
  // stage + swizzle: element i -> (frag, lane, j)
  for (int i = tid; i < NTILES * KSTEPS * 32 * 16; i += 256) {
    const int j  = i & 15;
    const int ln = (i >> 4) & 31;
    const int f  = i >> 9;                         // frag = nTile*KSTEPS + kstep
    const int kstep = f % KSTEPS;
    const int nT    = f / KSTEPS;
    const int n  = nT * 16 + (ln & 15);
    const int kb = (ln >> 4) * 8;
    const int k  = kstep * 32 + kb + ((j < 8) ? j : (8 + j));  // j>=8 -> +16+(j-8)
    Wsw[i] = (_Float16)W[k * OUT + n];
  }
  __syncthreads();

  const int wave = tid >> 5;
  const int lane = tid & 31;
  const int rowTile = blockIdx.x * 8 + wave;
  const int rowBase = rowTile * 16;
  if (rowBase >= N) return;                        // wave-uniform guard

  const int mrow = rowBase + (lane & 15);
  const int kb   = (lane >> 4) * 8;
  const int nl   = lane & 15;

  // load + convert all A fragments once (reused across every column tile)
  v16h aF[KSTEPS];
#pragma unroll
  for (int ks = 0; ks < KSTEPS; ++ks) {
    const float* ap = A + (size_t)mrow * DH + ks * 32 + kb;
    const float4 p0 = *(const float4*)(ap);
    const float4 p1 = *(const float4*)(ap + 4);
    const float4 p2 = *(const float4*)(ap + 16);
    const float4 p3 = *(const float4*)(ap + 20);
    aF[ks][0] = (_Float16)p0.x; aF[ks][1] = (_Float16)p0.y;
    aF[ks][2] = (_Float16)p0.z; aF[ks][3] = (_Float16)p0.w;
    aF[ks][4] = (_Float16)p1.x; aF[ks][5] = (_Float16)p1.y;
    aF[ks][6] = (_Float16)p1.z; aF[ks][7] = (_Float16)p1.w;
    aF[ks][8]  = (_Float16)p2.x; aF[ks][9]  = (_Float16)p2.y;
    aF[ks][10] = (_Float16)p2.z; aF[ks][11] = (_Float16)p2.w;
    aF[ks][12] = (_Float16)p3.x; aF[ks][13] = (_Float16)p3.y;
    aF[ks][14] = (_Float16)p3.z; aF[ks][15] = (_Float16)p3.w;
  }

  for (int nT = 0; nT < NTILES; ++nT) {
    v8f acc = {};
#pragma unroll
    for (int ks = 0; ks < KSTEPS; ++ks) {
      const v16h b = *(const v16h*)&Wsw[((nT * KSTEPS + ks) * 32 + lane) * 16];
      acc = __builtin_amdgcn_wmma_f32_16x16x32_f16(
          false, aF[ks], false, b, (short)0, acc, false, false);
    }
    const int dRow = rowBase + (lane >> 4) * 8;
#pragma unroll
    for (int v = 0; v < 8; ++v)
      C[(size_t)(dRow + v) * OUT + nT * 16 + nl] = acc[v];
  }
}

// ---------------- agg init: agg = h * dis^2 + b   (float4 streaming) ----------------
template <int D>
__global__ void self_init_k(const float* __restrict__ h, const float* __restrict__ dis,
                            const float* __restrict__ b, float* __restrict__ agg, int N) {
  constexpr int DV = D / 4;
  const float4* h4 = (const float4*)h;
  const float4* b4 = (const float4*)b;
  float4* a4 = (float4*)agg;
  size_t i = (size_t)blockIdx.x * blockDim.x + threadIdx.x;
  size_t total = (size_t)N * DV;
  size_t stride = (size_t)gridDim.x * blockDim.x;
  for (; i < total; i += stride) {
    const int row = (int)(i / DV);
    const int c4  = (int)(i % DV);
    const float d2 = dis[row] * dis[row];
    const float4 hv = h4[i];
    const float4 bv = b4[c4];
    float4 r;
    r.x = hv.x * d2 + bv.x; r.y = hv.y * d2 + bv.y;
    r.z = hv.z * d2 + bv.z; r.w = hv.w * d2 + bv.w;
    a4[i] = r;
  }
}

// ---------------- edge scatter-add: one wave32 per edge, float4 gather ----------------
template <int D>
__global__ void edge_agg_k(const int* __restrict__ src, const int* __restrict__ dst,
                           const float* __restrict__ dis, const float* __restrict__ h,
                           float* __restrict__ agg, int E) {
  constexpr int DV = D / 4;                        // float4 lanes needed per edge
  const int lane = threadIdx.x & 31;
  int w = (blockIdx.x * blockDim.x + threadIdx.x) >> 5;
  const int nw = (gridDim.x * blockDim.x) >> 5;
  for (; w < E; w += nw) {
    const int s = src[w];
    const int d = dst[w];
    const float nrm = dis[s] * dis[d];
    if (lane < DV) {
      const float4 v = ((const float4*)(h + (size_t)s * D))[lane];
      float* ad = agg + (size_t)d * D + lane * 4;
      atomicAdd(ad + 0, v.x * nrm);
      atomicAdd(ad + 1, v.y * nrm);
      atomicAdd(ad + 2, v.z * nrm);
      atomicAdd(ad + 3, v.w * nrm);
    }
  }
}

// ---------------- BatchNorm column stats: stats[c]=sum, stats[128+c]=sumsq ----------------
__global__ void bn_stats_k(const float* __restrict__ h, float* __restrict__ stats, int N) {
  const int col = threadIdx.x;   // blockDim = 128
  float s = 0.0f, s2 = 0.0f;
  for (int r = blockIdx.x; r < N; r += gridDim.x) {
    float v = h[(size_t)r * DH + col];
    s += v;
    s2 += v * v;
  }
  atomicAdd(&stats[col], s);
  atomicAdd(&stats[DH + col], s2);
}

// fold mean/var/gamma/beta into per-column scale/shift (in place in stats)
__global__ void bn_finalize_k(float* __restrict__ stats, const float* __restrict__ g,
                              const float* __restrict__ be, int N) {
  const int c = threadIdx.x;     // blockDim = 128, single block
  const float invN = 1.0f / (float)N;
  const float mu = stats[c] * invN;
  const float var = stats[DH + c] * invN - mu * mu;
  const float sc = rsqrtf(var + BN_EPS) * g[c];
  stats[c] = sc;
  stats[DH + c] = be[c] - mu * sc;
}

// ---------------- BN apply + ReLU (float4 streaming) ----------------
__global__ void bn_relu_k(const float* __restrict__ agg, const float* __restrict__ stats,
                          float* __restrict__ out, int N) {
  const float4* a4 = (const float4*)agg;
  const float4* sc4 = (const float4*)stats;            // [0..31]  scale
  const float4* sh4 = (const float4*)(stats + DH);     // [0..31]  shift
  float4* o4 = (float4*)out;
  size_t i = (size_t)blockIdx.x * blockDim.x + threadIdx.x;
  size_t total = (size_t)N * (DH / 4);
  size_t stride = (size_t)gridDim.x * blockDim.x;
  for (; i < total; i += stride) {
    const int c4 = (int)(i & (DH / 4 - 1));
    const float4 a = a4[i];
    const float4 sc = sc4[c4];
    const float4 sh = sh4[c4];
    float4 r;
    r.x = fmaxf(a.x * sc.x + sh.x, 0.0f);
    r.y = fmaxf(a.y * sc.y + sh.y, 0.0f);
    r.z = fmaxf(a.z * sc.z + sh.z, 0.0f);
    r.w = fmaxf(a.w * sc.w + sh.w, 0.0f);
    o4[i] = r;
  }
}

// ---------------- host-side orchestration ----------------
extern "C" void kernel_launch(void* const* d_in, const int* in_sizes, int n_in,
                              void* d_out, int out_size, void* d_ws, size_t ws_size,
                              hipStream_t stream) {
  const float* x  = (const float*)d_in[0];
  const int*   ei = (const int*)d_in[1];         // [2, E] row-major
  const int*   src = ei;
  const int*   dst = ei + NEDGES;
  const float* W1 = (const float*)d_in[2];  const float* b1 = (const float*)d_in[3];
  const float* W2 = (const float*)d_in[4];  const float* b2 = (const float*)d_in[5];
  const float* W3 = (const float*)d_in[6];  const float* b3 = (const float*)d_in[7];
  const float* W4 = (const float*)d_in[8];  const float* b4 = (const float*)d_in[9];
  const float* g1 = (const float*)d_in[10]; const float* be1 = (const float*)d_in[11];
  const float* g2 = (const float*)d_in[12]; const float* be2 = (const float*)d_in[13];
  const float* g3 = (const float*)d_in[14]; const float* be3 = (const float*)d_in[15];
  float* out = (float*)d_out;

  // workspace carve-up (floats)
  float* dis   = (float*)d_ws;                         // N
  float* bufA  = dis + NNODES;                         // N*128
  float* bufB  = bufA + (size_t)NNODES * DH;           // N*128
  float* stats = bufB + (size_t)NNODES * DH;           // 256

  const int TB = 256;
  const int gElem = 4096;
  const int gEdge = 16384;
  const int gGemm = (NNODES / 16 + 7) / 8;

  // degree -> dis
  zero_k<<<1024, TB, 0, stream>>>(dis, NNODES);
  degree_k<<<2048, TB, 0, stream>>>(dst, dis, NEDGES);
  rsqrt_k<<<512, TB, 0, stream>>>(dis, NNODES);

  // ---- layer 1: x -> bufA (gemm), agg in bufB, act back to bufA
  gemm_wmma_k<DH><<<gGemm, TB, 0, stream>>>(x, W1, bufA, NNODES);
  self_init_k<DH><<<gElem, TB, 0, stream>>>(bufA, dis, b1, bufB, NNODES);
  edge_agg_k<DH><<<gEdge, TB, 0, stream>>>(src, dst, dis, bufA, bufB, NEDGES);
  zero_k<<<1, TB, 0, stream>>>(stats, 2 * DH);
  bn_stats_k<<<1024, DH, 0, stream>>>(bufB, stats, NNODES);
  bn_finalize_k<<<1, DH, 0, stream>>>(stats, g1, be1, NNODES);
  bn_relu_k<<<gElem, TB, 0, stream>>>(bufB, stats, bufA, NNODES);

  // ---- layer 2: bufA -> bufB (gemm), agg in bufA, act to bufB
  gemm_wmma_k<DH><<<gGemm, TB, 0, stream>>>(bufA, W2, bufB, NNODES);
  self_init_k<DH><<<gElem, TB, 0, stream>>>(bufB, dis, b2, bufA, NNODES);
  edge_agg_k<DH><<<gEdge, TB, 0, stream>>>(src, dst, dis, bufB, bufA, NEDGES);
  zero_k<<<1, TB, 0, stream>>>(stats, 2 * DH);
  bn_stats_k<<<1024, DH, 0, stream>>>(bufA, stats, NNODES);
  bn_finalize_k<<<1, DH, 0, stream>>>(stats, g2, be2, NNODES);
  bn_relu_k<<<gElem, TB, 0, stream>>>(bufA, stats, bufB, NNODES);

  // ---- layer 3: bufB -> bufA (gemm), agg in bufB, act to bufA
  gemm_wmma_k<DH><<<gGemm, TB, 0, stream>>>(bufB, W3, bufA, NNODES);
  self_init_k<DH><<<gElem, TB, 0, stream>>>(bufA, dis, b3, bufB, NNODES);
  edge_agg_k<DH><<<gEdge, TB, 0, stream>>>(src, dst, dis, bufA, bufB, NEDGES);
  zero_k<<<1, TB, 0, stream>>>(stats, 2 * DH);
  bn_stats_k<<<1024, DH, 0, stream>>>(bufB, stats, NNODES);
  bn_finalize_k<<<1, DH, 0, stream>>>(stats, g3, be3, NNODES);
  bn_relu_k<<<gElem, TB, 0, stream>>>(bufB, stats, bufA, NNODES);

  // ---- layer 4 (OUT=16): bufA -> bufB[:N*16] (gemm), aggregate into d_out
  gemm_wmma_k<NCLS><<<gGemm, TB, 0, stream>>>(bufA, W4, bufB, NNODES);
  self_init_k<NCLS><<<gElem, TB, 0, stream>>>(bufB, dis, b4, out, NNODES);
  edge_agg_k<NCLS><<<gEdge, TB, 0, stream>>>(src, dst, dis, bufB, out, NEDGES);

  (void)in_sizes; (void)n_in; (void)out_size; (void)ws_size;
}
